// DVAE_48765058678983
// MI455X (gfx1250) — matrix operand order repaired and checked by reading, not compiled, not run on hardware
//
#include <hip/hip_runtime.h>
#include <hip/hip_bf16.h>
#include <stdint.h>
#include <math.h>

// ============================================================================
// CDNA5 (gfx1250) DVAE forward.  Heavy math on v_wmma_f32_16x16x32_bf16,
// weight tiles streamed to LDS with global_load_async_to_lds_b128 (ASYNCcnt),
// fragments loaded from LDS as ds_load_b128.
// d_in flatten assumption: d_in[0] = x; then params leaves in jax pytree order
// (dict keys sorted alphabetically, lists in index order), cp -> (b,w),
// npm -> (b,g).
// ============================================================================

typedef __attribute__((ext_vector_type(16))) __bf16 v16bf;
typedef __attribute__((ext_vector_type(8)))  float  v8f;

#define BM 128
#define BN 64
#define BK 32
#define LDA 40    // A tile row stride (__bf16): 80 B, 16-B aligned
#define LDBT 40   // B tile (n-major) row stride (__bf16): 80 B, 16-B aligned

union UFrag { uint4 q[2]; v16bf v; };

// ---- WMMA fragment loaders (ISA 7.12.2 layouts, wave32), b128 LDS reads ----
// A[16x32]: lane l holds row m=l&15; halves h0..7 -> k=kq+h, h8..15 -> k=16+kq+h
// with kq = (l>>4)*8.  Contiguous in LDS row-major [m][k].
__device__ __forceinline__ v16bf frag_a(const __bf16* As, int mbase, int lane) {
  const __bf16* p = As + (mbase + (lane & 15)) * LDA + ((lane >> 4) << 3);
  UFrag u;
  u.q[0] = *(const uint4*)p;          // k = kq .. kq+7
  u.q[1] = *(const uint4*)(p + 16);   // k = kq+16 .. kq+23
  return u.v;
}
// B[32x16]: lane l holds col n=l&15; halves h0..7 -> k=kb+h, h8..15 -> k=16+kb+h, kb = (l>>4)*8.
// Contiguous when B tile stored n-major [n][k].
__device__ __forceinline__ v16bf frag_b(const __bf16* Bt, int nbase, int lane) {
  const __bf16* p = Bt + (nbase + (lane & 15)) * LDBT + ((lane >> 4) << 3);
  UFrag u;
  u.q[0] = *(const uint4*)p;          // k = kb .. kb+7
  u.q[1] = *(const uint4*)(p + 16);   // k = kb+16 .. kb+23
  return u.v;
}

// ---- CDNA5 async global->LDS 16-byte copy (tracked by ASYNCcnt) ------------
__device__ __forceinline__ void async_copy16(void* lds_dst, const void* gsrc) {
  uint32_t l = (uint32_t)(uintptr_t)lds_dst;
  uint64_t g = (uint64_t)(uintptr_t)gsrc;
  asm volatile("global_load_async_to_lds_b128 %0, %1, off"
               :: "v"(l), "v"(g) : "memory");
}
__device__ __forceinline__ void async_wait0() {
  asm volatile("s_wait_asynccnt 0x0" ::: "memory");
}

#define WMMA_BF16(A, B, Cacc) \
  __builtin_amdgcn_wmma_f32_16x16x32_bf16(false, (A), false, (B), (short)0, (Cacc), false, false)

// ---- implicit-GEMM conv: out[n,m] = sum_k im2col(x)[m,k] * wN[n,k] ---------
// wN: bf16, padded [Np][Kp] (Kp multiple of 32, Np multiple of 64, zero-filled)
__global__ __launch_bounds__(256)
void k_conv_wmma(const float* __restrict__ x, const __bf16* __restrict__ wN,
                 const float* __restrict__ bias, const float* __restrict__ resid,
                 float* __restrict__ out,
                 int Cin, int H, int W, int Ho, int Wo, int Cout,
                 int ks, int pad, int stride, int Kp)
{
  const int M = Ho * Wo, N = Cout, K = Cin * ks * ks;
  const int m0 = blockIdx.x * BM, n0 = blockIdx.y * BN;
  __shared__ __align__(16) __bf16 As[BM * LDA];
  __shared__ __align__(16) __bf16 Bt[BN * LDBT];
  const int t = threadIdx.x;
  const int lane = t & 31, wave = t >> 5;
  const int wm = (wave & 3) * 32, wn = (wave >> 2) * 32;
  const int kk2 = ks * ks;
  // B async-copy coordinates: thread -> (n, 8-element k-chunk)
  const int bn = t >> 2, bch = (t & 3) << 3;
  v8f acc[2][2] = {};

  for (int k0 = 0; k0 < K; k0 += BK) {
    // B tile: one async b128 per thread, contiguous padded weights
    async_copy16(&Bt[bn * LDBT + bch],
                 wN + (size_t)(n0 + bn) * Kp + (k0 + bch));
#pragma unroll
    for (int e = 0; e < 16; ++e) {              // A tile: 128x32 gather + cvt
      int idx = t + (e << 8);
      int r = idx >> 5, cc = idx & 31;
      int m = m0 + r, k = k0 + cc;
      float v = 0.0f;
      if (m < M && k < K) {
        int ci = k / kk2, rem = k - ci * kk2;
        int ky = rem / ks, kx = rem - ky * ks;
        int oy = m / Wo, ox = m - oy * Wo;
        int iy = oy * stride + ky - pad;
        int ix = ox * stride + kx - pad;
        if ((unsigned)iy < (unsigned)H && (unsigned)ix < (unsigned)W)
          v = x[(size_t)ci * H * W + (size_t)iy * W + ix];
      }
      As[r * LDA + cc] = (__bf16)v;
    }
    if (k0 + BK < K)
      __builtin_prefetch(wN + (size_t)n0 * Kp + (k0 + BK), 0, 0);
    async_wait0();
    __syncthreads();

    v16bf a0 = frag_a(As, wm, lane), a1 = frag_a(As, wm + 16, lane);
    v16bf b0 = frag_b(Bt, wn, lane), b1 = frag_b(Bt, wn + 16, lane);
    acc[0][0] = WMMA_BF16(a0, b0, acc[0][0]);
    acc[0][1] = WMMA_BF16(a0, b1, acc[0][1]);
    acc[1][0] = WMMA_BF16(a1, b0, acc[1][0]);
    acc[1][1] = WMMA_BF16(a1, b1, acc[1][1]);
    __syncthreads();
  }

  const int lm = lane >> 4, ln = lane & 15;
#pragma unroll
  for (int i = 0; i < 2; ++i)
#pragma unroll
    for (int j = 0; j < 2; ++j)
#pragma unroll
      for (int r = 0; r < 8; ++r) {
        int m = m0 + wm + i * 16 + lm * 8 + r;
        int n = n0 + wn + j * 16 + ln;
        if (m < M && n < N) {
          float v = acc[i][j][r];
          if (bias) v += bias[n];
          size_t oo = (size_t)n * M + m;        // NCHW feature-major
          if (resid) v += resid[oo];
          out[oo] = v;
        }
      }
}

// ---- generic strided GEMM (attention, VQ distances) ------------------------
__global__ __launch_bounds__(256)
void k_gemm_wmma(const float* __restrict__ A, const float* __restrict__ Bm,
                 const float* __restrict__ bias, const float* __restrict__ resid,
                 float* __restrict__ out, int M, int N, int K,
                 long long Am, long long Ak, long long Bk, long long Bn,
                 long long Om, long long On)
{
  const int m0 = blockIdx.x * BM, n0 = blockIdx.y * BN;
  __shared__ __align__(16) __bf16 As[BM * LDA];
  __shared__ __align__(16) __bf16 Bt[BN * LDBT];
  const int t = threadIdx.x;
  const int lane = t & 31, wave = t >> 5;
  const int wm = (wave & 3) * 32, wn = (wave >> 2) * 32;
  v8f acc[2][2] = {};

  for (int k0 = 0; k0 < K; k0 += BK) {
#pragma unroll
    for (int e = 0; e < 16; ++e) {
      int idx = t + (e << 8);
      int r = idx >> 5, cc = idx & 31;
      int m = m0 + r, k = k0 + cc;
      float v = 0.0f;
      if (m < M && k < K) v = A[(long long)m * Am + (long long)k * Ak];
      As[r * LDA + cc] = (__bf16)v;
    }
#pragma unroll
    for (int e = 0; e < 8; ++e) {
      int idx = t + (e << 8);
      int n = idx & 63, r = idx >> 6;           // n contiguous across threads
      int k = k0 + r, nn = n0 + n;
      float v = 0.0f;
      if (k < K && nn < N) v = Bm[(long long)k * Bk + (long long)nn * Bn];
      Bt[n * LDBT + r] = (__bf16)v;             // store n-major
    }
    __syncthreads();
    v16bf a0 = frag_a(As, wm, lane), a1 = frag_a(As, wm + 16, lane);
    v16bf b0 = frag_b(Bt, wn, lane), b1 = frag_b(Bt, wn + 16, lane);
    acc[0][0] = WMMA_BF16(a0, b0, acc[0][0]);
    acc[0][1] = WMMA_BF16(a0, b1, acc[0][1]);
    acc[1][0] = WMMA_BF16(a1, b0, acc[1][0]);
    acc[1][1] = WMMA_BF16(a1, b1, acc[1][1]);
    __syncthreads();
  }

  const int lm = lane >> 4, ln = lane & 15;
#pragma unroll
  for (int i = 0; i < 2; ++i)
#pragma unroll
    for (int j = 0; j < 2; ++j)
#pragma unroll
      for (int r = 0; r < 8; ++r) {
        int m = m0 + wm + i * 16 + lm * 8 + r;
        int n = n0 + wn + j * 16 + ln;
        if (m < M && n < N) {
          float v = acc[i][j][r];
          if (bias) v += bias[n];
          long long oo = (long long)m * Om + (long long)n * On;
          if (resid) v += resid[oo];
          out[oo] = v;
        }
      }
}

// ---- weight prep: OIHW f32 [N][K] -> bf16 padded [Np][Kp], zero-filled -----
__global__ void k_wprep(const float* __restrict__ w, __bf16* __restrict__ wN,
                        int K, int N, int Kp, int Np) {
  int i = blockIdx.x * 256 + threadIdx.x;
  if (i < Np * Kp) {
    int n = i / Kp, k = i - n * Kp;
    float v = (n < N && k < K) ? w[(size_t)n * K + k] : 0.0f;
    wN[i] = (__bf16)v;
  }
}

// ---- group norm ------------------------------------------------------------
__global__ __launch_bounds__(256)
void k_gnstat(const float* __restrict__ x, float* __restrict__ mv, int C, int HW) {
  __shared__ float sh[256];
  int g = blockIdx.x, t = threadIdx.x;
  int cg = C / 32;
  size_t base = (size_t)g * cg * HW;
  size_t n = (size_t)cg * HW;
  float s = 0.f, ss = 0.f;
  for (size_t i = t; i < n; i += 256) { float v = x[base + i]; s += v; ss += v * v; }
  sh[t] = s; __syncthreads();
  for (int k = 128; k > 0; k >>= 1) { if (t < k) sh[t] += sh[t + k]; __syncthreads(); }
  s = sh[0]; __syncthreads();
  sh[t] = ss; __syncthreads();
  for (int k = 128; k > 0; k >>= 1) { if (t < k) sh[t] += sh[t + k]; __syncthreads(); }
  ss = sh[0];
  if (t == 0) {
    float m = s / (float)n;
    float var = ss / (float)n - m * m;
    mv[2 * g] = m; mv[2 * g + 1] = rsqrtf(var + 1e-6f);
  }
}

__global__ void k_gnapply(const float* __restrict__ x, const float* __restrict__ mv,
                          const float* __restrict__ gamma, const float* __restrict__ beta,
                          float* __restrict__ y, int C, int HW, int swish) {
  int i = blockIdx.x * 256 + threadIdx.x;
  if (i < C * HW) {
    int c = i / HW;
    int g = c / (C / 32);
    float v = (x[i] - mv[2 * g]) * mv[2 * g + 1] * gamma[c] + beta[c];
    if (swish) v = v / (1.0f + __expf(-v));
    y[i] = v;
  }
}

// ---- softmax over rows (scale applied pre-softmax) -------------------------
__global__ __launch_bounds__(256)
void k_softmax(float* __restrict__ att, int L, float scale) {
  __shared__ float sh[256];
  int row = blockIdx.x, t = threadIdx.x;
  float* p = att + (size_t)row * L;
  float mx = -3.4e38f;
  for (int i = t; i < L; i += 256) mx = fmaxf(mx, p[i] * scale);
  sh[t] = mx; __syncthreads();
  for (int k = 128; k > 0; k >>= 1) { if (t < k) sh[t] = fmaxf(sh[t], sh[t + k]); __syncthreads(); }
  mx = sh[0]; __syncthreads();
  float sum = 0.f;
  for (int i = t; i < L; i += 256) { float e = __expf(p[i] * scale - mx); p[i] = e; sum += e; }
  sh[t] = sum; __syncthreads();
  for (int k = 128; k > 0; k >>= 1) { if (t < k) sh[t] += sh[t + k]; __syncthreads(); }
  float inv = 1.0f / sh[0];
  for (int i = t; i < L; i += 256) p[i] *= inv;
}

// ---- entropy map -----------------------------------------------------------
__global__ void k_gray(const float* __restrict__ x, float* __restrict__ g) {
  int i = blockIdx.x * 256 + threadIdx.x;
  if (i < 65536) g[i] = (x[i] + x[65536 + i] + x[131072 + i]) * (1.0f / 3.0f);
}

__global__ __launch_bounds__(256)
void k_entropy(const float* __restrict__ gray, float* __restrict__ ent) {
  __shared__ float pix[256];
  __shared__ float sh[256];
  int p = blockIdx.x, t = threadIdx.x;
  int py = p >> 4, px = p & 15;
  int ty = t >> 4, tx = t & 15;
  pix[t] = gray[(py * 16 + ty) * 256 + (px * 16 + tx)];
  __syncthreads();
  float bin = (float)t * (1.0f / 255.0f);
  float s = 0.0f;
#pragma unroll 4
  for (int i = 0; i < 256; ++i) {
    float d = (pix[i] - bin) * 100.0f;          // / sigma(0.01)
    s += __expf(-0.5f * d * d);
  }
  float pdf = s * (1.0f / 256.0f);
  sh[t] = pdf; __syncthreads();
  for (int k = 128; k > 0; k >>= 1) { if (t < k) sh[t] += sh[t + k]; __syncthreads(); }
  float tot = sh[0]; __syncthreads();
  float q = fmaxf(pdf / (tot + 1e-10f), 1e-10f);
  sh[t] = -q * __log2f(q); __syncthreads();
  for (int k = 128; k > 0; k >>= 1) { if (t < k) sh[t] += sh[t + k]; __syncthreads(); }
  if (t == 0) ent[p] = sh[0];
}

// ---- quantile(0.5, linear) via bitonic sort + grain mask -------------------
__global__ __launch_bounds__(256)
void k_quant_grain(const float* __restrict__ ent, float* __restrict__ grain) {
  __shared__ float s[256];
  __shared__ float thr;
  int t = threadIdx.x;
  s[t] = ent[t]; __syncthreads();
  for (int kk = 2; kk <= 256; kk <<= 1)
    for (int j = kk >> 1; j > 0; j >>= 1) {
      int ixj = t ^ j;
      if (ixj > t) {
        bool up = ((t & kk) == 0);
        float a = s[t], b = s[ixj];
        if ((a > b) == up) { s[t] = b; s[ixj] = a; }
      }
      __syncthreads();
    }
  if (t == 0) thr = 0.5f * (s[127] + s[128]);
  __syncthreads();
  grain[t] = (ent[t] > thr) ? 1.0f : 0.0f;
}

// ---- VQ --------------------------------------------------------------------
__global__ void k_cn2(const float* __restrict__ cb, float* __restrict__ cn2) {
  int n = blockIdx.x * 256 + threadIdx.x;
  if (n < 1024) {
    float s = 0.f;
    for (int d = 0; d < 256; ++d) { float v = cb[(size_t)n * 256 + d]; s += v * v; }
    cn2[n] = s;
  }
}

__global__ __launch_bounds__(256)
void k_argmin(const float* __restrict__ dots, const float* __restrict__ cn2,
              int* __restrict__ idx, int Ncode) {
  __shared__ float sv[256];
  __shared__ int   si[256];
  int m = blockIdx.x, t = threadIdx.x;
  const float* d = dots + (size_t)m * Ncode;
  float best = 3.4e38f; int bi = 0;
  for (int i = t; i < Ncode; i += 256) {
    float v = cn2[i] - 2.0f * d[i];
    if (v < best) { best = v; bi = i; }
  }
  sv[t] = best; si[t] = bi; __syncthreads();
  for (int k = 128; k > 0; k >>= 1) {
    if (t < k && sv[t + k] < sv[t]) { sv[t] = sv[t + k]; si[t] = si[t + k]; }
    __syncthreads();
  }
  if (t == 0) idx[m] = si[0];
}

__global__ void k_vqgather(const float* __restrict__ cb, const int* __restrict__ idx,
                           float* __restrict__ zq, int D, int hw) {
  int i = blockIdx.x * 256 + threadIdx.x;
  if (i < D * hw) {
    int d = i / hw, m = i - d * hw;
    zq[i] = cb[(size_t)idx[m] * D + d];
  }
}

// ---- routing + upsample ----------------------------------------------------
__global__ void k_route(const float* __restrict__ qf, const float* __restrict__ qc,
                        const float* __restrict__ grain, float* __restrict__ z,
                        int C, int Hf, int Wf) {
  int i = blockIdx.x * 256 + threadIdx.x;
  if (i < C * Hf * Wf) {
    int c = i / (Hf * Wf), rem = i - c * Hf * Wf;
    int y = rem / Wf, x = rem - y * Wf;
    float g = grain[(y >> 1) * (Wf >> 1) + (x >> 1)];
    float cu = qc[(size_t)c * (Hf / 2) * (Wf / 2) + (y >> 1) * (Wf / 2) + (x >> 1)];
    z[i] = g * qf[i] + (1.0f - g) * cu;
  }
}

__global__ void k_up2(const float* __restrict__ in, float* __restrict__ out,
                      int C, int H, int W) {
  int i = blockIdx.x * 256 + threadIdx.x;
  int H2 = 2 * H, W2 = 2 * W;
  if (i < C * H2 * W2) {
    int c = i / (H2 * W2), rem = i - c * H2 * W2;
    int y = rem / W2, x = rem - y * W2;
    out[i] = in[(size_t)c * H * W + (y >> 1) * W + (x >> 1)];
  }
}

// ============================================================================
// Host orchestration
// ============================================================================
struct Conv { const float* w; const float* b; int cin, cout, ks; };
struct Norm { const float* g; const float* b; };
struct Res  { Conv c1, c2, nin; Norm n1, n2; bool has_nin; };
struct Attn { Conv q, k, v, o; Norm n; };

struct Walker {
  void* const* din; int i;
  const float* next() { return (const float*)din[i++]; }
};
static Conv getConv(Walker& w, int cin, int cout, int ks) {
  Conv c; c.b = w.next(); c.w = w.next(); c.cin = cin; c.cout = cout; c.ks = ks; return c;
}
static Norm getNorm(Walker& w) { Norm n; n.b = w.next(); n.g = w.next(); return n; }
static Res getRes(Walker& w, int cin, int cout) {
  Res r; r.c1 = getConv(w, cin, cout, 3); r.c2 = getConv(w, cout, cout, 3);
  r.n1 = getNorm(w); r.n2 = getNorm(w);
  r.has_nin = (cin != cout);
  if (r.has_nin) r.nin = getConv(w, cin, cout, 1); else r.nin = Conv{};
  return r;
}
static Attn getAttn(Walker& w, int c) {  // alphabetical: k,n,o,q,v
  Attn a; a.k = getConv(w, c, c, 1); a.n = getNorm(w); a.o = getConv(w, c, c, 1);
  a.q = getConv(w, c, c, 1); a.v = getConv(w, c, c, 1); return a;
}

struct Ctx {
  hipStream_t s;
  float *B0, *B1, *T0, *T1, *T2;
  float *AQ, *AK, *AV, *ATT;
  __bf16* WB;
  float* MV;
  float *GRAY, *ENT, *GRAIN;
  float *HF, *ZFi, *ZCo, *QF, *QC, *ZR;
  float* CN2; int* IDX;
  const float* cb;
};

static inline int cdiv(int a, int b) { return (a + b - 1) / b; }

static void conv2d(Ctx& c, const Conv& cv, const float* x, float* out,
                   int H, int W, int Ho, int Wo, int pad, int stride, const float* resid) {
  int K = cv.cin * cv.ks * cv.ks, N = cv.cout, M = Ho * Wo;
  int Kp = (K + 31) & ~31, Np = (N + 63) & ~63;
  k_wprep<<<cdiv(Kp * Np, 256), 256, 0, c.s>>>(cv.w, c.WB, K, N, Kp, Np);
  dim3 grid(cdiv(M, BM), cdiv(N, BN));
  k_conv_wmma<<<grid, 256, 0, c.s>>>(x, c.WB, cv.b, resid, out,
                                     cv.cin, H, W, Ho, Wo, N, cv.ks, pad, stride, Kp);
}

static void gnorm(Ctx& c, const Norm& nm, const float* x, float* y, int C, int HW, bool swish) {
  k_gnstat<<<32, 256, 0, c.s>>>(x, c.MV, C, HW);
  k_gnapply<<<cdiv(C * HW, 256), 256, 0, c.s>>>(x, c.MV, nm.g, nm.b, y, C, HW, swish ? 1 : 0);
}

static void resblock(Ctx& c, const Res& r, const float* x, float* out, int H, int W) {
  int cin = r.c1.cin, cout = r.c1.cout, HW = H * W;
  gnorm(c, r.n1, x, c.T0, cin, HW, true);
  conv2d(c, r.c1, c.T0, c.T1, H, W, H, W, 1, 1, nullptr);
  gnorm(c, r.n2, c.T1, c.T0, cout, HW, true);
  const float* resid = x;
  if (r.has_nin) { conv2d(c, r.nin, x, c.T2, H, W, H, W, 0, 1, nullptr); resid = c.T2; }
  conv2d(c, r.c2, c.T0, out, H, W, H, W, 1, 1, resid);
}

static void attnblock(Ctx& c, const Attn& a, const float* x, float* out, int H, int W) {
  int C = a.q.cin, HW = H * W;
  gnorm(c, a.n, x, c.T0, C, HW, false);
  conv2d(c, a.q, c.T0, c.AQ, H, W, H, W, 0, 1, nullptr);
  conv2d(c, a.k, c.T0, c.AK, H, W, H, W, 0, 1, nullptr);
  conv2d(c, a.v, c.T0, c.AV, H, W, H, W, 0, 1, nullptr);
  // att[m,n] = sum_k q[k,m] * kf[k,n]
  dim3 g1(cdiv(HW, BM), cdiv(HW, BN));
  k_gemm_wmma<<<g1, 256, 0, c.s>>>(c.AQ, c.AK, nullptr, nullptr, c.ATT,
      HW, HW, C, 1LL, (long long)HW, (long long)HW, 1LL, (long long)HW, 1LL);
  k_softmax<<<HW, 256, 0, c.s>>>(c.ATT, HW, 1.0f / sqrtf((float)C));
  // h[n,m] = sum_k att[m,k] * v[n,k]
  dim3 g2(cdiv(HW, BM), cdiv(C, BN));
  k_gemm_wmma<<<g2, 256, 0, c.s>>>(c.ATT, c.AV, nullptr, nullptr, c.T0,
      HW, C, HW, (long long)HW, 1LL, 1LL, (long long)HW, 1LL, (long long)HW);
  conv2d(c, a.o, c.T0, out, H, W, H, W, 0, 1, x);
}

static void vq_lookup(Ctx& c, const float* z, float* zq, int hw) {
  // dots[m,n] = sum_d z[d,m] * cb[n,d]
  dim3 g(cdiv(hw, BM), cdiv(1024, BN));
  k_gemm_wmma<<<g, 256, 0, c.s>>>(z, c.cb, nullptr, nullptr, c.ATT,
      hw, 1024, 256, 1LL, (long long)hw, 1LL, 256LL, 1024LL, 1LL);
  k_argmin<<<hw, 256, 0, c.s>>>(c.ATT, c.CN2, c.IDX, 1024);
  k_vqgather<<<cdiv(256 * hw, 256), 256, 0, c.s>>>(c.cb, c.IDX, zq, 256, hw);
}

extern "C" void kernel_launch(void* const* d_in, const int* in_sizes, int n_in,
                              void* d_out, int out_size, void* d_ws, size_t ws_size,
                              hipStream_t stream) {
  (void)in_sizes; (void)n_in; (void)out_size; (void)ws_size;

  // ---- workspace layout ----
  char* base = (char*)d_ws;
  size_t off = 0;
  auto alloc = [&](size_t b) -> char* {
    char* p = base + off;
    off = (off + b + 255) & ~(size_t)255;
    return p;
  };
  const size_t BIG = (size_t)256 * 65536 * 4;   // 256ch @ 256x256 f32
  const size_t MED = (size_t)128 * 65536 * 4;
  Ctx C;
  C.s = stream;
  C.B0 = (float*)alloc(BIG);  C.B1 = (float*)alloc(BIG);  C.T0 = (float*)alloc(BIG);
  C.T1 = (float*)alloc(MED);  C.T2 = (float*)alloc(MED);
  C.AQ = (float*)alloc((size_t)512 * 1024 * 4);
  C.AK = (float*)alloc((size_t)512 * 1024 * 4);
  C.AV = (float*)alloc((size_t)512 * 1024 * 4);
  C.ATT = (float*)alloc((size_t)1024 * 1024 * 4);
  C.WB = (__bf16*)alloc((size_t)4608 * 512 * 2 + 256);
  C.MV = (float*)alloc(64 * 4);
  C.GRAY = (float*)alloc(65536 * 4);
  C.ENT = (float*)alloc(256 * 4);
  C.GRAIN = (float*)alloc(256 * 4);
  C.HF = (float*)alloc((size_t)256 * 1024 * 4);
  C.ZFi = (float*)alloc((size_t)256 * 1024 * 4);
  C.ZCo = (float*)alloc((size_t)256 * 256 * 4);
  C.QF  = (float*)alloc((size_t)256 * 1024 * 4);
  C.QC  = (float*)alloc((size_t)256 * 256 * 4);
  C.ZR  = (float*)alloc((size_t)256 * 1024 * 4);
  C.CN2 = (float*)alloc(1024 * 4);
  C.IDX = (int*)alloc(1024 * 4);

  const float* x = (const float*)d_in[0];

  // ---- parameter walk (jax pytree order) ----
  Walker w{d_in, 1};
  C.cb = w.next();                               // codebook [1024,256]

  // decoder: conv_in, conv_out, mid{a,b1,b2}, norm_out, up[0..3]{attn,block,up}
  Conv d_conv_in  = getConv(w, 256, 512, 3);
  Conv d_conv_out = getConv(w, 128, 3, 3);
  Attn d_mid_a    = getAttn(w, 512);
  Res  d_mid_b1   = getRes(w, 512, 512);
  Res  d_mid_b2   = getRes(w, 512, 512);
  Norm d_norm_out = getNorm(w);
  struct DLvl { int na; Attn a[3]; Res b[3]; bool up_; Conv up; } dlv[4];
  dlv[0].na = 0; dlv[0].b[0] = getRes(w, 256, 128); dlv[0].b[1] = getRes(w, 128, 128);
  dlv[0].b[2] = getRes(w, 128, 128); dlv[0].up_ = false;
  dlv[1].na = 0; dlv[1].b[0] = getRes(w, 256, 256); dlv[1].b[1] = getRes(w, 256, 256);
  dlv[1].b[2] = getRes(w, 256, 256); dlv[1].up_ = true; dlv[1].up = getConv(w, 256, 256, 3);
  dlv[2].na = 0; dlv[2].b[0] = getRes(w, 512, 256); dlv[2].b[1] = getRes(w, 256, 256);
  dlv[2].b[2] = getRes(w, 256, 256); dlv[2].up_ = true; dlv[2].up = getConv(w, 256, 256, 3);
  dlv[3].na = 3; dlv[3].a[0] = getAttn(w, 512); dlv[3].a[1] = getAttn(w, 512);
  dlv[3].a[2] = getAttn(w, 512);
  dlv[3].b[0] = getRes(w, 512, 512); dlv[3].b[1] = getRes(w, 512, 512);
  dlv[3].b[2] = getRes(w, 512, 512); dlv[3].up_ = true; dlv[3].up = getConv(w, 512, 512, 3);

  // encoder: conv_in, down[0..4]{attn,block,down}, mid_c, mid_f, norm_c, norm_f, out_c, out_f
  Conv e_conv_in = getConv(w, 3, 128, 3);
  struct ELvl { int na; Attn a[2]; Res b[2]; bool down_; Conv down; } elv[5];
  elv[0].na = 0; elv[0].b[0] = getRes(w, 128, 128); elv[0].b[1] = getRes(w, 128, 128);
  elv[0].down_ = true; elv[0].down = getConv(w, 128, 128, 3);
  elv[1].na = 0; elv[1].b[0] = getRes(w, 128, 128); elv[1].b[1] = getRes(w, 128, 128);
  elv[1].down_ = true; elv[1].down = getConv(w, 128, 128, 3);
  elv[2].na = 0; elv[2].b[0] = getRes(w, 128, 256); elv[2].b[1] = getRes(w, 256, 256);
  elv[2].down_ = true; elv[2].down = getConv(w, 256, 256, 3);
  elv[3].na = 2; elv[3].a[0] = getAttn(w, 256); elv[3].a[1] = getAttn(w, 256);
  elv[3].b[0] = getRes(w, 256, 256); elv[3].b[1] = getRes(w, 256, 256);
  elv[3].down_ = true; elv[3].down = getConv(w, 256, 256, 3);
  elv[4].na = 2; elv[4].a[0] = getAttn(w, 512); elv[4].a[1] = getAttn(w, 512);
  elv[4].b[0] = getRes(w, 256, 512); elv[4].b[1] = getRes(w, 512, 512);
  elv[4].down_ = false;
  Attn e_midc_a  = getAttn(w, 512);
  Res  e_midc_b1 = getRes(w, 512, 512);
  Res  e_midc_b2 = getRes(w, 512, 512);
  Attn e_midf_a  = getAttn(w, 256);
  Res  e_midf_b1 = getRes(w, 256, 256);
  Res  e_midf_b2 = getRes(w, 256, 256);
  Norm e_norm_c  = getNorm(w);
  Norm e_norm_f  = getNorm(w);
  Conv e_out_c   = getConv(w, 512, 256, 3);
  Conv e_out_f   = getConv(w, 256, 256, 3);

  // ---- entropy routing mask + codebook norms ----
  k_gray<<<256, 256, 0, stream>>>(x, C.GRAY);
  k_entropy<<<256, 256, 0, stream>>>(C.GRAY, C.ENT);
  k_quant_grain<<<1, 256, 0, stream>>>(C.ENT, C.GRAIN);
  k_cn2<<<4, 256, 0, stream>>>(C.cb, C.CN2);

  // ---- encoder ----
  float* cur = C.B0; float* nxt = C.B1;
  conv2d(C, e_conv_in, x, cur, 256, 256, 256, 256, 1, 1, nullptr);
  int H = 256;
  for (int i = 0; i < 5; ++i) {
    for (int j = 0; j < 2; ++j) {
      resblock(C, elv[i].b[j], cur, nxt, H, H); { float* t = cur; cur = nxt; nxt = t; }
      if (elv[i].na) { attnblock(C, elv[i].a[j], cur, nxt, H, H); float* t = cur; cur = nxt; nxt = t; }
    }
    if (i == 3)
      (void)hipMemcpyAsync(C.HF, cur, (size_t)256 * 32 * 32 * 4, hipMemcpyDeviceToDevice, stream);
    if (elv[i].down_) {
      conv2d(C, elv[i].down, cur, nxt, H, H, H / 2, H / 2, 0, 2, nullptr);
      float* t = cur; cur = nxt; nxt = t; H /= 2;
    }
  }
  // coarse head (512ch @ 16x16)
  resblock(C, e_midc_b1, cur, nxt, 16, 16); { float* t = cur; cur = nxt; nxt = t; }
  attnblock(C, e_midc_a, cur, nxt, 16, 16); { float* t = cur; cur = nxt; nxt = t; }
  resblock(C, e_midc_b2, cur, nxt, 16, 16); { float* t = cur; cur = nxt; nxt = t; }
  gnorm(C, e_norm_c, cur, C.T0, 512, 256, true);
  conv2d(C, e_out_c, C.T0, C.ZCo, 16, 16, 16, 16, 1, 1, nullptr);
  // fine head (256ch @ 32x32)
  (void)hipMemcpyAsync(cur, C.HF, (size_t)256 * 32 * 32 * 4, hipMemcpyDeviceToDevice, stream);
  resblock(C, e_midf_b1, cur, nxt, 32, 32); { float* t = cur; cur = nxt; nxt = t; }
  attnblock(C, e_midf_a, cur, nxt, 32, 32); { float* t = cur; cur = nxt; nxt = t; }
  resblock(C, e_midf_b2, cur, nxt, 32, 32); { float* t = cur; cur = nxt; nxt = t; }
  gnorm(C, e_norm_f, cur, C.T0, 256, 1024, true);
  conv2d(C, e_out_f, C.T0, C.ZFi, 32, 32, 32, 32, 1, 1, nullptr);

  // ---- VQ + routing ----
  vq_lookup(C, C.ZFi, C.QF, 1024);
  vq_lookup(C, C.ZCo, C.QC, 256);
  k_route<<<cdiv(256 * 1024, 256), 256, 0, stream>>>(C.QF, C.QC, C.GRAIN, C.ZR, 256, 32, 32);

  // ---- decoder ----
  conv2d(C, d_conv_in, C.ZR, cur, 32, 32, 32, 32, 1, 1, nullptr);
  resblock(C, d_mid_b1, cur, nxt, 32, 32); { float* t = cur; cur = nxt; nxt = t; }
  attnblock(C, d_mid_a, cur, nxt, 32, 32); { float* t = cur; cur = nxt; nxt = t; }
  resblock(C, d_mid_b2, cur, nxt, 32, 32); { float* t = cur; cur = nxt; nxt = t; }
  H = 32;
  for (int i = 3; i >= 0; --i) {
    for (int j = 0; j < 3; ++j) {
      resblock(C, dlv[i].b[j], cur, nxt, H, H); { float* t = cur; cur = nxt; nxt = t; }
      if (dlv[i].na) { attnblock(C, dlv[i].a[j], cur, nxt, H, H); float* t = cur; cur = nxt; nxt = t; }
    }
    if (dlv[i].up_) {
      int Cc = dlv[i].b[0].c1.cout;
      k_up2<<<cdiv(Cc * 4 * H * H, 256), 256, 0, stream>>>(cur, nxt, Cc, H, H);
      { float* t = cur; cur = nxt; nxt = t; } H *= 2;
      conv2d(C, dlv[i].up, cur, nxt, H, H, H, H, 1, 1, nullptr);
      { float* t = cur; cur = nxt; nxt = t; }
    }
  }
  gnorm(C, d_norm_out, cur, C.T0, 128, 65536, true);
  conv2d(C, d_conv_out, C.T0, (float*)d_out, 256, 256, 256, 256, 1, 1, nullptr);
}
